// MetaAttention_2525440770086
// MI455X (gfx1250) — matrix operand order repaired and checked by reading, not compiled, not run on hardware
//
#include <hip/hip_runtime.h>

// MI455X / gfx1250, wave32. Multi-head attention:
//   q,k,v = x@W{q,k,v}.T ; flash softmax(q k^T / sqrt(64)) v ; out = merge @ Wo.T
// Matmuls via v_wmma_f32_16x16x32_f16. All shared operand tiles (GEMM B-tiles,
// attention K/V tiles) staged into LDS with the Tensor Data Mover
// (tensor_load_to_lds / TENSORcnt), double-buffered to overlap DMA and WMMA.

typedef __attribute__((ext_vector_type(16))) _Float16 v16h;
typedef __attribute__((ext_vector_type(8)))  float    v8f;
typedef __attribute__((ext_vector_type(4)))  unsigned int v4u;
typedef __attribute__((ext_vector_type(8)))  int      v8i_t;
typedef __attribute__((ext_vector_type(4)))  int      v4i_t;

union Frag16 { v16h h; unsigned int u[8]; };

// A-matrix (16x32 f16) K offset for dword v (0..7), lane half g (0/1).
// ISA 7.12.2: lanes0-15 V0..3 -> K0..7, V4..7 -> K16..23; lanes16-31 +8.
__device__ __forceinline__ int ka_off(int g, int v) {
    return ((v < 4) ? (2 * v) : (16 + 2 * (v - 4))) + 8 * g;
}
// B-matrix (32x16 f16) K offset: same K order as A (lanes0-15 V0..3 -> K0..7,
// V4..7 -> K16..23; lanes16-31 +8).
__device__ __forceinline__ int kb_off(int g, int v) {
    return ((v < 4) ? (2 * v) : (16 + 2 * (v - 4))) + 8 * g;
}

// XOR lane shuffle via ds_swizzle (group-of-32 mode, and_mask=0x1f).
template <int MASK>
__device__ __forceinline__ float swz_xor(float x) {
    int i = __builtin_bit_cast(int, x);
    i = __builtin_amdgcn_ds_swizzle(i, (MASK << 10) | 0x1f);
    return __builtin_bit_cast(float, i);
}
__device__ __forceinline__ float rowmax16(float x) {
    x = fmaxf(x, swz_xor<1>(x));
    x = fmaxf(x, swz_xor<2>(x));
    x = fmaxf(x, swz_xor<4>(x));
    x = fmaxf(x, swz_xor<8>(x));
    return x;
}
__device__ __forceinline__ float rowsum16(float x) {
    x += swz_xor<1>(x);
    x += swz_xor<2>(x);
    x += swz_xor<4>(x);
    x += swz_xor<8>(x);
    return x;
}

// Tensor Data Mover: DMA a [tile_rows x tile_k] f16 tile (row stride `stride`
// elements) from global into LDS at byte offset lds_off (packed row-major).
// D# layout per CDNA5 ISA ch.8: group0 = {count, lds_addr, global_addr, type=2},
// group1 = {data_size=2B, tensor_dim0/1, tile_dim0/1, tensor_dim0_stride}.
__device__ __forceinline__ void tdm_load_tile(unsigned int lds_off,
                                              const _Float16* gsrc,
                                              int tile_k, int tile_rows,
                                              int tdim0, int tdim1,
                                              long long stride) {
    unsigned long long ga = (unsigned long long)(uintptr_t)gsrc;
    v4u g0;
    g0[0] = 1u;                                    // count=1 (valid user D#)
    g0[1] = lds_off;                               // LDS byte address
    g0[2] = (unsigned int)ga;                      // global addr [31:0]
    g0[3] = (unsigned int)(ga >> 32) | (2u << 30); // addr [56:32] | type=2
    v8i_t g1;
    g1[0] = 0x00010000;                            // data_size=1 -> 2 bytes
    g1[1] = (int)(((unsigned)tdim0 & 0xffffu) << 16);
    g1[2] = (int)((((unsigned)tdim0 >> 16) & 0xffffu) |
                  (((unsigned)tdim1 & 0xffffu) << 16));
    g1[3] = (int)((((unsigned)tdim1 >> 16) & 0xffffu) |
                  ((unsigned)tile_k << 16));       // tile_dim0
    g1[4] = (int)((unsigned)tile_rows & 0xffffu);  // tile_dim1 (tile_dim2=0)
    g1[5] = (int)(unsigned int)stride;             // tensor_dim0_stride[31:0]
    g1[6] = (int)(unsigned int)((stride >> 32) & 0xffff); // stride[47:32]
    g1[7] = 0;                                     // tensor_dim1_stride unused (2D)
    v4i_t gz = {0, 0, 0, 0};
#if defined(__clang_major__) && (__clang_major__ >= 23)
    v8i_t gz8 = {0, 0, 0, 0, 0, 0, 0, 0};
    __builtin_amdgcn_tensor_load_to_lds(g0, g1, gz, gz, gz8, 0);
#else
    __builtin_amdgcn_tensor_load_to_lds(g0, g1, gz, gz, 0);
#endif
}

__global__ void cvt_f32_f16(const float* __restrict__ in, _Float16* __restrict__ out, int n) {
    int i = blockIdx.x * blockDim.x + threadIdx.x;
    if (i < n) out[i] = (_Float16)in[i];
}

// C = A[M x 768] * Bw[N x 768]^T  (y = x @ W.T), block tile 128M x 64N,
// 8 waves x (16M x 64N). B tile (64 cols x 32 K halves, 4KB) staged in LDS by
// the TDM, double buffered; A fragments loaded directly from global.
// mode 0: f16 Q (scaled 0.125) -> [BH,1024,64]
// mode 1: f16 K                -> [BH,1024,64]
// mode 2: f16 V transposed     -> [BH,64,1024]
// mode 3: f32 plain            -> [M,768]
__global__ __launch_bounds__(256) void gemm16(const _Float16* __restrict__ A,
                                              const _Float16* __restrict__ Bw,
                                              void* __restrict__ Out, int mode) {
    __shared__ _Float16 sB[2][64 * 32];  // 2 x 4KB double buffer
    const int K = 768;
    const int NSTEP = K / 32;  // 24
    const int lane = threadIdx.x & 31;
    const int w    = threadIdx.x >> 5;
    const int g    = lane >> 4;
    const int ln   = lane & 15;
    const int row0 = blockIdx.x * 128 + w * 16;
    const int col0 = blockIdx.y * 64;
    const int arow = row0 + ln;

    v8f acc[4];
#pragma unroll
    for (int nf = 0; nf < 4; ++nf)
#pragma unroll
        for (int r = 0; r < 8; ++r) acc[nf][r] = 0.0f;

    // prologue: stage B tile for k0=0
    if (w == 0) {
        tdm_load_tile((unsigned int)(uintptr_t)&sB[0][0], Bw + col0 * K,
                      32, 64, K, 768, K);
    }

    for (int s = 0; s < NSTEP; ++s) {
        const int k0 = s * 32;
        if (w == 0) {
            if (s + 1 < NSTEP) {
                tdm_load_tile((unsigned int)(uintptr_t)&sB[(s + 1) & 1][0],
                              Bw + col0 * K + (s + 1) * 32, 32, 64, K, 768, K);
                __builtin_amdgcn_s_wait_tensorcnt((short)1);  // tile s landed
            } else {
                __builtin_amdgcn_s_wait_tensorcnt((short)0);
            }
        }
        __syncthreads();  // B tile s visible to all waves

        // prefetch A stream a couple of K-steps ahead
        __builtin_prefetch((const char*)(A + arow * K + k0) + 256, 0, 1);

        Frag16 a;
#pragma unroll
        for (int v = 0; v < 8; ++v)
            a.u[v] = *(const unsigned int*)(A + arow * K + k0 + ka_off(g, v));

        const _Float16* bt = sB[s & 1];
#pragma unroll
        for (int nf = 0; nf < 4; ++nf) {
            const int bcol = nf * 16 + ln;
            Frag16 b;
#pragma unroll
            for (int v = 0; v < 8; ++v)
                b.u[v] = *(const unsigned int*)(bt + bcol * 32 + kb_off(g, v));
            acc[nf] = __builtin_amdgcn_wmma_f32_16x16x32_f16(
                false, a.h, false, b.h, (short)0, acc[nf], false, false);
        }
        __syncthreads();  // done reading tile s before it is overwritten
    }

#pragma unroll
    for (int nf = 0; nf < 4; ++nf) {
        const int gcol = col0 + nf * 16 + ln;
#pragma unroll
        for (int r = 0; r < 8; ++r) {
            const int grow = row0 + r + 8 * g;
            const float val = acc[nf][r];
            if (mode == 3) {
                ((float*)Out)[grow * 768 + gcol] = val;
            } else {
                const int b  = grow >> 10, nq = grow & 1023;
                const int h  = gcol >> 6,  d  = gcol & 63;
                const int bh = b * 12 + h;
                _Float16* O = (_Float16*)Out;
                if (mode == 0)      O[(bh * 1024 + nq) * 64 + d] = (_Float16)(val * 0.125f);
                else if (mode == 1) O[(bh * 1024 + nq) * 64 + d] = (_Float16)val;
                else                O[(bh * 64 + d) * 1024 + nq] = (_Float16)val;
            }
        }
    }
}

// Flash attention per (bh, 128-query tile); each wave owns 16 queries.
// Q,K: [BH,1024,64] f16 (Q pre-scaled).  Vt: [BH,64,1024] f16.
// K and V 32-key tiles are TDM-staged into LDS (shared by all 8 waves),
// double-buffered. Vec out: [B*1024, 768] f16 (heads merged).
__global__ __launch_bounds__(256) void attn16(const _Float16* __restrict__ Q,
                                              const _Float16* __restrict__ Kd,
                                              const _Float16* __restrict__ Vt,
                                              _Float16* __restrict__ Vec) {
    __shared__ _Float16 sK[2][32 * 64];    // [key][dh], 2 x 4KB
    __shared__ _Float16 sV[2][64 * 32];    // [dh][key], 2 x 4KB
    __shared__ unsigned int sPmem[8][256]; // per-wave 16x32 f16 P tile (1KB each)
    const int lane = threadIdx.x & 31;
    const int w    = threadIdx.x >> 5;
    const int g    = lane >> 4;
    const int ln   = lane & 15;
    const int bh   = blockIdx.y;
    const int b    = bh / 12, h = bh % 12;
    const int q0   = blockIdx.x * 128 + w * 16;
    const int qrow = q0 + ln;

    Frag16 qa[2];
#pragma unroll
    for (int kc = 0; kc < 2; ++kc)
#pragma unroll
        for (int v = 0; v < 8; ++v)
            qa[kc].u[v] = *(const unsigned int*)(Q + (bh * 1024 + qrow) * 64 + kc * 32 + ka_off(g, v));

    float m[8], l[8];
    v8f o[4];
#pragma unroll
    for (int r = 0; r < 8; ++r) { m[r] = -1e30f; l[r] = 0.0f; }
#pragma unroll
    for (int c = 0; c < 4; ++c)
#pragma unroll
        for (int r = 0; r < 8; ++r) o[c][r] = 0.0f;

    _Float16* sp = (_Float16*)sPmem[w];

    // prologue: stage K/V tiles for keys [0,32)
    if (w == 0) {
        tdm_load_tile((unsigned int)(uintptr_t)&sK[0][0], Kd + bh * 1024 * 64,
                      64, 32, 64, 1024, 64);
        tdm_load_tile((unsigned int)(uintptr_t)&sV[0][0], Vt + bh * 64 * 1024,
                      32, 64, 1024, 64, 1024);
    }

    for (int s = 0; s < 32; ++s) {
        const int j0 = s * 32;
        if (w == 0) {
            if (s + 1 < 32) {
                const int j1 = (s + 1) * 32;
                tdm_load_tile((unsigned int)(uintptr_t)&sK[(s + 1) & 1][0],
                              Kd + (bh * 1024 + j1) * 64, 64, 32, 64, 1024, 64);
                tdm_load_tile((unsigned int)(uintptr_t)&sV[(s + 1) & 1][0],
                              Vt + bh * 64 * 1024 + j1, 32, 64, 1024, 64, 1024);
                __builtin_amdgcn_s_wait_tensorcnt((short)2);  // step-s tiles landed
            } else {
                __builtin_amdgcn_s_wait_tensorcnt((short)0);
            }
        }
        __syncthreads();  // K/V tiles for step s visible to all waves
        const _Float16* kt = sK[s & 1];
        const _Float16* vt = sV[s & 1];

        // scores S[16 queries x 32 keys] = Q * K^T (two 16-key halves)
        v8f sc[2];
#pragma unroll
        for (int kh = 0; kh < 2; ++kh)
#pragma unroll
            for (int r = 0; r < 8; ++r) sc[kh][r] = 0.0f;
#pragma unroll
        for (int kh = 0; kh < 2; ++kh) {
            const int kloc = kh * 16 + ln;  // key within tile
#pragma unroll
            for (int kc = 0; kc < 2; ++kc) {
                Frag16 kb;
#pragma unroll
                for (int v = 0; v < 8; ++v)
                    kb.u[v] = *(const unsigned int*)(kt + kloc * 64 + kc * 32 + kb_off(g, v));
                sc[kh] = __builtin_amdgcn_wmma_f32_16x16x32_f16(
                    false, qa[kc].h, false, kb.h, (short)0, sc[kh], false, false);
            }
        }
        // online softmax: rows live in (vgpr r -> row r+8g), cols across 16 lanes
        float alpha[8], p0[8], p1[8];
#pragma unroll
        for (int r = 0; r < 8; ++r) {
            float mx = rowmax16(fmaxf(sc[0][r], sc[1][r]));
            const float mn = fmaxf(m[r], mx);
            alpha[r] = __expf(m[r] - mn);
            m[r] = mn;
            p0[r] = __expf(sc[0][r] - mn);
            p1[r] = __expf(sc[1][r] - mn);
            const float t = rowsum16(p0[r] + p1[r]);
            l[r] = l[r] * alpha[r] + t;
        }
#pragma unroll
        for (int c = 0; c < 4; ++c)
#pragma unroll
            for (int r = 0; r < 8; ++r) o[c][r] *= alpha[r];

        // C-layout P (f32) -> f16 -> LDS -> reread as A-layout 16x32 fragment
#pragma unroll
        for (int r = 0; r < 8; ++r) {
            const int prow = r + 8 * g;
            sp[prow * 32 + ln]      = (_Float16)p0[r];
            sp[prow * 32 + 16 + ln] = (_Float16)p1[r];
        }
        asm volatile("s_wait_dscnt 0x0" ::: "memory");
        Frag16 pa;
#pragma unroll
        for (int v = 0; v < 8; ++v)
            pa.u[v] = *(const unsigned int*)(sp + ln * 32 + ka_off(g, v));

        // O += P * V  (V tile is [dh][key]: contiguous in key)
#pragma unroll
        for (int c = 0; c < 4; ++c) {
            const int d = c * 16 + ln;
            Frag16 vb;
#pragma unroll
            for (int v = 0; v < 8; ++v)
                vb.u[v] = *(const unsigned int*)(vt + d * 32 + kb_off(g, v));
            o[c] = __builtin_amdgcn_wmma_f32_16x16x32_f16(
                false, pa.h, false, vb.h, (short)0, o[c], false, false);
        }
        __syncthreads();  // done reading step-s tiles before overwrite
    }

#pragma unroll
    for (int c = 0; c < 4; ++c) {
        const int d = c * 16 + ln;
#pragma unroll
        for (int r = 0; r < 8; ++r) {
            const int grow = q0 + r + 8 * g;
            Vec[(b * 1024 + grow) * 768 + h * 64 + d] = (_Float16)(o[c][r] / l[r]);
        }
    }
}

extern "C" void kernel_launch(void* const* d_in, const int* in_sizes, int n_in,
                              void* d_out, int out_size, void* d_ws, size_t ws_size,
                              hipStream_t stream) {
    (void)in_sizes; (void)n_in; (void)out_size; (void)ws_size;
    const float* x  = (const float*)d_in[0];
    const float* Wq = (const float*)d_in[1];
    const float* Wk = (const float*)d_in[2];
    const float* Wv = (const float*)d_in[3];
    const float* Wo = (const float*)d_in[4];

    const int NX = 8 * 1024 * 768;  // 6291456
    const int NW = 768 * 768;       // 589824

    _Float16* xh  = (_Float16*)d_ws;
    _Float16* wqh = xh + NX;
    _Float16* wkh = wqh + NW;
    _Float16* wvh = wkh + NW;
    _Float16* woh = wvh + NW;
    _Float16* qws = woh + NW;       // [96,1024,64]
    _Float16* kws = qws + NX;       // [96,1024,64]
    _Float16* vtw = kws + NX;       // [96,64,1024]
    _Float16* vec = vtw + NX;       // [8192,768]

    cvt_f32_f16<<<(NX + 255) / 256, 256, 0, stream>>>(x,  xh,  NX);
    cvt_f32_f16<<<(NW + 255) / 256, 256, 0, stream>>>(Wq, wqh, NW);
    cvt_f32_f16<<<(NW + 255) / 256, 256, 0, stream>>>(Wk, wkh, NW);
    cvt_f32_f16<<<(NW + 255) / 256, 256, 0, stream>>>(Wv, wvh, NW);
    cvt_f32_f16<<<(NW + 255) / 256, 256, 0, stream>>>(Wo, woh, NW);

    dim3 gg(64, 12);  // 8192/128 x 768/64
    gemm16<<<gg, 256, 0, stream>>>(xh, wqh, qws, 0);
    gemm16<<<gg, 256, 0, stream>>>(xh, wkh, kws, 1);
    gemm16<<<gg, 256, 0, stream>>>(xh, wvh, vtw, 2);

    attn16<<<dim3(8, 96), 256, 0, stream>>>(qws, kws, vtw, vec);

    gemm16<<<gg, 256, 0, stream>>>(vec, woh, d_out, 3);
}
